// KANLayer_29996051595740
// MI455X (gfx1250) — matrix operand order repaired and checked by reading, not compiled, not run on hardware
//
#include <hip/hip_runtime.h>
#include <hip/hip_bf16.h>

typedef __attribute__((ext_vector_type(16))) _Float16 v16h;
typedef __attribute__((ext_vector_type(8)))  _Float16 v8h;
typedef __attribute__((ext_vector_type(8)))  float    v8f;

#define KB_B 4096
#define KB_N 32
#define KB_H 64
#define KB_O 32
#define KB_P 496   // N*(N-1)/2

// ---------------------------------------------------------------------------
// Kernel 1: per (b,n) compute phi MLP, then u = W1a*phi + psi_b1, v = W1b*phi
// (psi_w1 = [H, 2O]; W1a = cols 0..31 acts on phi_i, W1b = cols 32..63 on phi_j)
// Store u,v as f16 rows of 64 for direct WMMA A-fragment consumption.
// ---------------------------------------------------------------------------
__global__ __launch_bounds__(256) void kan_phi_uv(
    const float* __restrict__ x,
    const float* __restrict__ phi_w1, const float* __restrict__ phi_b1,
    const float* __restrict__ phi_w2, const float* __restrict__ phi_b2,
    const float* __restrict__ psi_w1, const float* __restrict__ psi_b1,
    _Float16* __restrict__ U, _Float16* __restrict__ V)
{
    __shared__ float sw[6368];
    float* sW1  = sw;            // 64   phi_w1[:,0]
    float* sB1  = sw + 64;       // 64   phi_b1
    float* sW2  = sw + 128;      // 2048 phi_w2 [32,64]
    float* sB2  = sw + 2176;     // 32   phi_b2
    float* sPW1 = sw + 2208;     // 4096 psi_w1 [64,64]
    float* sPB1 = sw + 6304;     // 64   psi_b1

    const int tid = threadIdx.x;
    for (int k = tid; k < 64;   k += 256) { sW1[k] = phi_w1[k]; sB1[k] = phi_b1[k]; sPB1[k] = psi_b1[k]; }
    for (int k = tid; k < 2048; k += 256) sW2[k]  = phi_w2[k];
    for (int k = tid; k < 32;   k += 256) sB2[k]  = phi_b2[k];
    for (int k = tid; k < 4096; k += 256) sPW1[k] = psi_w1[k];
    __syncthreads();

    const int row = blockIdx.x * 256 + tid;      // row = b*N + n  (< 131072)
    const float xv = x[row];

    float po[KB_O];
#pragma unroll
    for (int o = 0; o < KB_O; ++o) po[o] = sB2[o];

    for (int h = 0; h < KB_H; ++h) {
        float h1 = xv * sW1[h] + sB1[h];
        h1 = h1 > 0.f ? h1 : 0.f;
#pragma unroll
        for (int o = 0; o < KB_O; ++o) po[o] += h1 * sW2[o * KB_H + h];
    }

    _Float16* urow = U + (size_t)row * KB_H;
    _Float16* vrow = V + (size_t)row * KB_H;
    for (int h = 0; h < KB_H; ++h) {
        float su = sPB1[h];      // fold psi_b1 into u
        float sv = 0.f;
        const float* wrow = sPW1 + h * (2 * KB_O);
#pragma unroll
        for (int o = 0; o < KB_O; ++o) {
            su += wrow[o]        * po[o];
            sv += wrow[KB_O + o] * po[o];
        }
        urow[h] = (_Float16)su;
        vrow[h] = (_Float16)sv;
    }
}

// ---------------------------------------------------------------------------
// Kernel 2: per 16-batch tile, all (i<j) pairs:
//   h2[16x64] = relu(u_i + v_j) (f16),  C[16x32] += h2 @ psi_w2^T  via WMMA
// 16 waves/block; wave w owns rows i=w and i=30-w (32 pairs each, wave 15: 16)
// with CONTIGUOUS j. A asm-barrier pins the two A fragments in distinct
// register octets so the allocator cannot serialize them through one octet
// (which caused WMMA WAR-hazard v_nops); WMMAs alternate a0/a1 so both stay
// live across the group. 4 accumulators keep adjacent WMMA dests distinct.
// ---------------------------------------------------------------------------
__global__ __launch_bounds__(512) void kan_psi_pairs(
    const _Float16* __restrict__ U, const _Float16* __restrict__ V,
    const float* __restrict__ psi_w2, const float* __restrict__ psi_b2,
    float* __restrict__ out)
{
    __shared__ float red[16][512];               // 32 KB

    const int tid  = threadIdx.x;
    const int w    = tid >> 5;                   // 0..15
    const int lane = tid & 31;
    const int base = blockIdx.x * 16;            // batch tile base

    // --- B fragments: psi_w2 [O,H] as f16, B[k=h][n=o], 2 N-tiles x 2 K-blocks.
    // 16-bit B 32x16 layout: lanes 0-15 hold K=0..7,16..23 (lane = N col),
    // lanes 16-31 hold K=8..15,24..31.
    const int col    = lane & 15;
    const int kbaseB = (lane < 16) ? 0 : 8;
    v16h Bf[2][2];
#pragma unroll
    for (int nt = 0; nt < 2; ++nt) {
        const float* wrow = psi_w2 + (size_t)(nt * 16 + col) * KB_H;
#pragma unroll
        for (int kb = 0; kb < 2; ++kb) {
            v16h b;
#pragma unroll
            for (int e = 0; e < 16; ++e) b[e] = (_Float16)wrow[kb * 32 + kbaseB + e + (e & 8)];
            Bf[nt][kb] = b;
        }
    }

    v8f c00 = {};   // N = 0..15,  K-block 0
    v8f c01 = {};   // N = 0..15,  K-block 1
    v8f c10 = {};   // N = 16..31, K-block 0
    v8f c11 = {};   // N = 16..31, K-block 1

    // --- A fragment addressing (16-bit A 16x32 layout):
    // lanes 0-15 : M=lane,    elems 0..7 -> K 0..7,   elems 8..15 -> K 16..23
    // lanes 16-31: M=lane-16, elems 0..7 -> K 8..15,  elems 8..15 -> K 24..31
    const int M    = lane & 15;
    const int ksel = (lane < 16) ? 0 : 8;
    const _Float16* ubase = U + ((size_t)(base + M) * KB_N) * KB_H;
    const _Float16* vbase = V + ((size_t)(base + M) * KB_N) * KB_H;

    auto do_row = [&](int i) {
        const _Float16* urow = ubase + i * KB_H;
        v8h u00 = *(const v8h*)(urow + ksel);
        v8h u01 = *(const v8h*)(urow + ksel + 16);
        v8h u10 = *(const v8h*)(urow + 32 + ksel);
        v8h u11 = *(const v8h*)(urow + 32 + ksel + 16);
#pragma unroll 2
        for (int j = i + 1; j < KB_N; ++j) {
            const _Float16* vrow = vbase + j * KB_H;
            v8h w00 = *(const v8h*)(vrow + ksel);
            v8h w01 = *(const v8h*)(vrow + ksel + 16);
            v8h w10 = *(const v8h*)(vrow + 32 + ksel);
            v8h w11 = *(const v8h*)(vrow + 32 + ksel + 16);

            v16h a0, a1;
#pragma unroll
            for (int e = 0; e < 8; ++e) {
                _Float16 s00 = (_Float16)(u00[e] + w00[e]);
                _Float16 s01 = (_Float16)(u01[e] + w01[e]);
                _Float16 s10 = (_Float16)(u10[e] + w10[e]);
                _Float16 s11 = (_Float16)(u11[e] + w11[e]);
                a0[e]     = s00 > (_Float16)0 ? s00 : (_Float16)0;
                a0[e + 8] = s01 > (_Float16)0 ? s01 : (_Float16)0;
                a1[e]     = s10 > (_Float16)0 ? s10 : (_Float16)0;
                a1[e + 8] = s11 > (_Float16)0 ? s11 : (_Float16)0;
            }
            // Pin a0 and a1 live simultaneously in distinct octets: prevents
            // the allocator from funneling both fragments through one octet
            // (that reuse forced WMMA WAR-hazard v_nops).
            asm volatile("" : "+v"(a0), "+v"(a1));
            // WMMAs alternate fragments; all destinations distinct.
            c00 = __builtin_amdgcn_wmma_f32_16x16x32_f16(false, a0, false, Bf[0][0], (short)0, c00, false, false);
            c01 = __builtin_amdgcn_wmma_f32_16x16x32_f16(false, a1, false, Bf[0][1], (short)0, c01, false, false);
            c10 = __builtin_amdgcn_wmma_f32_16x16x32_f16(false, a0, false, Bf[1][0], (short)0, c10, false, false);
            c11 = __builtin_amdgcn_wmma_f32_16x16x32_f16(false, a1, false, Bf[1][1], (short)0, c11, false, false);
        }
    };

    do_row(w);                    // i = 0..15
    if (w < 15) do_row(30 - w);   // i = 16..30 (wave 15 only owns i=15)

    // --- C layout: VGPR r, lanes 0-15 -> (M=r, N=lane); lanes 16-31 -> (M=r+8, N=lane-16)
    // Merge per-K-block partials while writing to LDS.
#pragma unroll
    for (int r = 0; r < 8; ++r) {
        const int Mr = (lane < 16) ? r : (r + 8);
        red[w][Mr * 32 + col]      = c00[r] + c01[r];
        red[w][Mr * 32 + 16 + col] = c10[r] + c11[r];
    }
    __syncthreads();

    {
        const int idx = tid;      // 512 threads, 512 elements
        float s = 0.f;
#pragma unroll
        for (int ww = 0; ww < 16; ++ww) s += red[ww][idx];
        const int Mi = idx >> 5;
        const int n  = idx & 31;
        out[(size_t)(base + Mi) * 32 + n] = s + (float)KB_P * psi_b2[n];
    }
}

// ---------------------------------------------------------------------------
extern "C" void kernel_launch(void* const* d_in, const int* in_sizes, int n_in,
                              void* d_out, int out_size, void* d_ws, size_t ws_size,
                              hipStream_t stream) {
    const float* x      = (const float*)d_in[0];
    const float* phi_w1 = (const float*)d_in[1];
    const float* phi_b1 = (const float*)d_in[2];
    const float* phi_w2 = (const float*)d_in[3];
    const float* phi_b2 = (const float*)d_in[4];
    const float* psi_w1 = (const float*)d_in[5];
    const float* psi_b1 = (const float*)d_in[6];
    const float* psi_w2 = (const float*)d_in[7];
    const float* psi_b2 = (const float*)d_in[8];

    _Float16* U = (_Float16*)d_ws;                       // [B*N][64] f16 = 16 MB
    _Float16* V = U + (size_t)KB_B * KB_N * KB_H;        // another 16 MB

    // Kernel 1: B*N = 131072 threads
    kan_phi_uv<<<(KB_B * KB_N) / 256, 256, 0, stream>>>(
        x, phi_w1, phi_b1, phi_w2, phi_b2, psi_w1, psi_b1, U, V);

    // Kernel 2: one block (16 waves) per 16-batch tile
    kan_psi_pairs<<<KB_B / 16, 512, 0, stream>>>(U, V, psi_w2, psi_b2, (float*)d_out);
}
